// RNNEncDec_36567351558878
// MI455X (gfx1250) — matrix operand order repaired and checked by reading, not compiled, not run on hardware
//
#include <hip/hip_runtime.h>
#include <cstdint>
#include <cstddef>

// ---------------------------------------------------------------------------
// RNN encoder-decoder (GRU seq2seq + maxout readout) for MI455X (gfx1250).
//
// Strategy: convert all GEMM operands to bf16 once per launch, batch all
// time-parallel GEMMs (input projections, y@V_o.T, the 768x500x30000 logits
// GEMM), and run the serial GRU recurrence as 2 fused WMMA kernels per step.
// All matrix math uses v_wmma_f32_16x16x32_bf16 (wave32).  Inner GEMM loops
// are software-pipelined (double-buffered fragments) so each WMMA group waits
// on loads issued one full k-iteration earlier instead of loadcnt==0.
// ---------------------------------------------------------------------------

typedef __attribute__((ext_vector_type(16))) __bf16 v16bf;
typedef __attribute__((ext_vector_type(8)))  float  v8f;
typedef unsigned int uintv4 __attribute__((ext_vector_type(4)));

union Frag32B { uintv4 q[2]; v16bf v; };

// A-matrix (16x32 bf16) per-lane k pattern: lanes 0-15 -> k in {0..7,16..23},
// lanes 16-31 -> k in {8..15,24..31}.  Two 16B loads per fragment.
__device__ __forceinline__ v16bf load_fragA(const unsigned short* rowk0, int lane) {
  const unsigned short* p = rowk0 + ((lane >> 4) << 3);   // +8 elems for upper half-wave
  Frag32B f;
  f.q[0] = *(const uintv4*)(p);        // k = base+0..7
  f.q[1] = *(const uintv4*)(p + 16);   // k = base+16..23
  return f.v;
}
// B-matrix (32x16 bf16): same k pattern as A: lanes 0-15 -> k in {0..7,16..23},
// lanes 16-31 -> k in {8..15,24..31}.
__device__ __forceinline__ v16bf load_fragB(const unsigned short* rowk0, int lane) {
  const unsigned short* p = rowk0 + ((lane >> 4) << 3);   // +8 elems for upper half-wave
  Frag32B f;
  f.q[0] = *(const uintv4*)(p);
  f.q[1] = *(const uintv4*)(p + 16);
  return f.v;
}
__device__ __forceinline__ v8f wmma_bf16(v16bf a, v16bf b, v8f c) {
  return __builtin_amdgcn_wmma_f32_16x16x32_bf16(false, a, false, b, (short)0, c,
                                                 false, false);
}
__device__ __forceinline__ unsigned short f2bf(float f) {   // RNE f32 -> bf16
  unsigned int u = __float_as_uint(f);
  u += 0x7fffu + ((u >> 16) & 1u);
  return (unsigned short)(u >> 16);
}
__device__ __forceinline__ float sigm(float x) { return 1.0f / (1.0f + __expf(-x)); }
__device__ __forceinline__ int imin(int a, int b) { return a < b ? a : b; }

// ---------------------------------------------------------------------------
// Generic GEMM: out[M,N] = A_bf16[M,Kp(lda)] @ W_bf16[Orows,Kp(ldw)]^T (+addC)
// Block = 128 threads = 4 waves; each wave computes a 16x64 tile (A fragment
// reused across 4 WMMAs); block tile = 32x128.  Double-buffered k-loop.
// mode 0: out[m*ldo+n];  mode 1 (logits): m = t*Bb+b -> out[(b*Tt+t)*ldo+n].
// ---------------------------------------------------------------------------
__global__ void gemm_bf16_k(const unsigned short* __restrict__ A, int lda,
                            const unsigned short* __restrict__ Wt, int ldw, int Orows,
                            float* __restrict__ out, int ldo, int M, int N, int Kp,
                            const float* __restrict__ addC, int mode, int Bb, int Tt)
{
  const int lane  = threadIdx.x & 31;
  const int w     = threadIdx.x >> 5;
  const int mbase = blockIdx.y * 32 + (w & 1) * 16;
  const int nbase = blockIdx.x * 128 + (w >> 1) * 64;
  const int arow  = imin(mbase + (lane & 15), M - 1);

  const unsigned short* Ap = A + (size_t)arow * lda;
  const unsigned short* Bp0 = Wt + (size_t)imin(nbase +  0 + (lane & 15), Orows - 1) * ldw;
  const unsigned short* Bp1 = Wt + (size_t)imin(nbase + 16 + (lane & 15), Orows - 1) * ldw;
  const unsigned short* Bp2 = Wt + (size_t)imin(nbase + 32 + (lane & 15), Orows - 1) * ldw;
  const unsigned short* Bp3 = Wt + (size_t)imin(nbase + 48 + (lane & 15), Orows - 1) * ldw;

  const v8f z8 = {0.f,0.f,0.f,0.f,0.f,0.f,0.f,0.f};
  v8f acc0 = z8, acc1 = z8, acc2 = z8, acc3 = z8;

  // software pipeline: prefetch k+32 fragments before consuming k fragments
  const int kLast = Kp - 32;
  v16bf aC  = load_fragA(Ap  + 0, lane);
  v16bf bC0 = load_fragB(Bp0 + 0, lane);
  v16bf bC1 = load_fragB(Bp1 + 0, lane);
  v16bf bC2 = load_fragB(Bp2 + 0, lane);
  v16bf bC3 = load_fragB(Bp3 + 0, lane);
  for (int k0 = 0; k0 < Kp; k0 += 32) {
    const int kn = (k0 + 32 < Kp) ? (k0 + 32) : kLast;   // clamped (branch-free)
    v16bf aN  = load_fragA(Ap  + kn, lane);
    v16bf bN0 = load_fragB(Bp0 + kn, lane);
    v16bf bN1 = load_fragB(Bp1 + kn, lane);
    v16bf bN2 = load_fragB(Bp2 + kn, lane);
    v16bf bN3 = load_fragB(Bp3 + kn, lane);
    acc0 = wmma_bf16(aC, bC0, acc0);
    acc1 = wmma_bf16(aC, bC1, acc1);
    acc2 = wmma_bf16(aC, bC2, acc2);
    acc3 = wmma_bf16(aC, bC3, acc3);
    aC = aN; bC0 = bN0; bC1 = bN1; bC2 = bN2; bC3 = bN3;
  }

  const int g = lane >> 4, nl = lane & 15;
  v8f acc[4] = {acc0, acc1, acc2, acc3};
#pragma unroll
  for (int j = 0; j < 4; ++j) {
    int n = nbase + j * 16 + nl;
    if (n >= N) continue;
#pragma unroll
    for (int e = 0; e < 8; ++e) {
      int m = mbase + e + 8 * g;
      if (m >= M) continue;
      float v = acc[j][e];
      if (addC) v += addC[(size_t)m * ldo + n];
      if (mode == 0) {
        out[(size_t)m * ldo + n] = v;
      } else {
        int tt = m / Bb, bb = m % Bb;
        out[((size_t)bb * Tt + tt) * ldo + n] = v;
      }
    }
  }
}

// ---------------------------------------------------------------------------
// GRU step kernel 1: r = sigmoid(projR + S@Ur^T + ctxR + br)  (z likewise)
// writes z (f32) and rh = bf16(r * S).   M = 32 rows fixed (B).
// Block = 256 threads = 8 waves over a 32x64 tile; grid.x covers H.
// ---------------------------------------------------------------------------
__global__ void step_rz_k(const unsigned short* __restrict__ Sbf,
                          const unsigned short* __restrict__ Ur,
                          const unsigned short* __restrict__ Uz,
                          int H, int Hp,
                          const float* __restrict__ projR, const float* __restrict__ projZ,
                          const float* __restrict__ ctxR,  const float* __restrict__ ctxZ,
                          const float* __restrict__ br,    const float* __restrict__ bz,
                          const float* __restrict__ Sf32,
                          unsigned short* __restrict__ rh, float* __restrict__ zout)
{
  const int lane  = threadIdx.x & 31;
  const int w     = threadIdx.x >> 5;
  const int mbase = (w & 1) * 16;
  const int nbase = blockIdx.x * 64 + (w >> 1) * 16;
  const int arow  = mbase + (lane & 15);
  const int brow  = imin(nbase + (lane & 15), H - 1);
  const unsigned short* Ap = Sbf + (size_t)arow * Hp;
  const unsigned short* Rp = Ur  + (size_t)brow * Hp;
  const unsigned short* Zp = Uz  + (size_t)brow * Hp;

  const v8f z8 = {0.f,0.f,0.f,0.f,0.f,0.f,0.f,0.f};
  v8f accR = z8, accZ = z8;

  const int kLast = Hp - 32;
  v16bf aC  = load_fragA(Ap + 0, lane);
  v16bf rC  = load_fragB(Rp + 0, lane);
  v16bf zC  = load_fragB(Zp + 0, lane);
  for (int k0 = 0; k0 < Hp; k0 += 32) {
    const int kn = (k0 + 32 < Hp) ? (k0 + 32) : kLast;
    v16bf aN = load_fragA(Ap + kn, lane);
    v16bf rN = load_fragB(Rp + kn, lane);
    v16bf zN = load_fragB(Zp + kn, lane);
    accR = wmma_bf16(aC, rC, accR);
    accZ = wmma_bf16(aC, zC, accZ);
    aC = aN; rC = rN; zC = zN;
  }

  const int g = lane >> 4, nc = nbase + (lane & 15);
  if (nc < H) {
    float bR = br[nc], bZ = bz[nc];
#pragma unroll
    for (int e = 0; e < 8; ++e) {
      int m = mbase + e + 8 * g;
      size_t i = (size_t)m * H + nc;
      float r = sigm(accR[e] + projR[i] + bR + (ctxR ? ctxR[i] : 0.0f));
      float z = sigm(accZ[e] + projZ[i] + bZ + (ctxZ ? ctxZ[i] : 0.0f));
      zout[i] = z;
      rh[(size_t)m * Hp + nc] = f2bf(r * Sf32[i]);
    }
  }
}

// ---------------------------------------------------------------------------
// GRU step kernel 2: cand = tanh(projC + RH@U^T + ctxC + b);
// h' = (1-z)h + z*cand; writes f32 state, bf16 state, optional s_all slot.
// ---------------------------------------------------------------------------
__global__ void step_upd_k(const unsigned short* __restrict__ RH,
                           const unsigned short* __restrict__ U,
                           int H, int Hp,
                           const float* __restrict__ projC, const float* __restrict__ ctxC,
                           const float* __restrict__ bc,
                           const float* __restrict__ zbuf,
                           float* __restrict__ Sf32,          // in/out (per-thread in-place)
                           unsigned short* __restrict__ Sbf,  // current bf16 state
                           unsigned short* __restrict__ SallDst) // nullable
{
  const int lane  = threadIdx.x & 31;
  const int w     = threadIdx.x >> 5;
  const int mbase = (w & 1) * 16;
  const int nbase = blockIdx.x * 64 + (w >> 1) * 16;
  const int arow  = mbase + (lane & 15);
  const int brow  = imin(nbase + (lane & 15), H - 1);
  const unsigned short* Ap = RH + (size_t)arow * Hp;
  const unsigned short* Up = U  + (size_t)brow * Hp;

  v8f acc = {0.f,0.f,0.f,0.f,0.f,0.f,0.f,0.f};
  const int kLast = Hp - 32;
  v16bf aC = load_fragA(Ap + 0, lane);
  v16bf bC = load_fragB(Up + 0, lane);
  for (int k0 = 0; k0 < Hp; k0 += 32) {
    const int kn = (k0 + 32 < Hp) ? (k0 + 32) : kLast;
    v16bf aN = load_fragA(Ap + kn, lane);
    v16bf bN = load_fragB(Up + kn, lane);
    acc = wmma_bf16(aC, bC, acc);
    aC = aN; bC = bN;
  }

  const int g = lane >> 4, nc = nbase + (lane & 15);
  if (nc < H) {
    float bC2 = bc[nc];
#pragma unroll
    for (int e = 0; e < 8; ++e) {
      int m = mbase + e + 8 * g;
      size_t i = (size_t)m * H + nc;
      float cand = tanhf(acc[e] + projC[i] + bC2 + (ctxC ? ctxC[i] : 0.0f));
      float z = zbuf[i];
      float hn = (1.0f - z) * Sf32[i] + z * cand;
      Sf32[i] = hn;
      unsigned short hb = f2bf(hn);
      Sbf[(size_t)m * Hp + nc] = hb;
      if (SallDst) SallDst[(size_t)m * Hp + nc] = hb;
    }
  }
}

// -------------------------- small helper kernels ---------------------------
__global__ void zero_u32_k(unsigned int* p, long n) {
  long i = (long)blockIdx.x * blockDim.x + threadIdx.x;
  long s = (long)gridDim.x * blockDim.x;
  for (; i < n; i += s) p[i] = 0u;
}

// row r = s*B + b ; token = tok[b*L + s] ; emits bf16 row with zero K-pad.
__global__ void gather_embed_k(const int* __restrict__ tok, const float* __restrict__ emb,
                               unsigned short* __restrict__ out, int Bn, int L, int E, int Ep) {
  int r = blockIdx.x;
  int s = r / Bn, b = r % Bn;
  int t = tok[b * L + s];
  const float* src = emb + (size_t)t * E;
  unsigned short* dst = out + (size_t)r * Ep;
  for (int e = threadIdx.x; e < Ep; e += blockDim.x)
    dst[e] = (e < E) ? f2bf(src[e]) : (unsigned short)0;
}

__global__ void cvt_bf16_k(const float* __restrict__ Wf, unsigned short* __restrict__ out,
                           int K, int Kp) {
  int o = blockIdx.x;
  const float* src = Wf + (size_t)o * K;
  unsigned short* dst = out + (size_t)o * Kp;
  for (int k = threadIdx.x; k < Kp; k += blockDim.x)
    dst[k] = (k < K) ? f2bf(src[k]) : (unsigned short)0;
}

__global__ void s0_tanh_k(const float* __restrict__ tmp, float* __restrict__ sf,
                          unsigned short* __restrict__ sbf, unsigned short* __restrict__ sall,
                          int H, int Hp) {
  int b = blockIdx.x;
  for (int n = threadIdx.x; n < H; n += blockDim.x) {
    float v = tanhf(tmp[(size_t)b * H + n]);
    sf[(size_t)b * H + n] = v;
    unsigned short u = f2bf(v);
    sbf[(size_t)b * Hp + n] = u;
    sall[(size_t)b * Hp + n] = u;
  }
}

__global__ void add_bcast_k(float* __restrict__ yv, const float* __restrict__ cCo,
                            int Bb, int Ncols) {
  int r = blockIdx.x, b = r % Bb;
  for (int n = threadIdx.x; n < Ncols; n += blockDim.x)
    yv[(size_t)r * Ncols + n] += cCo[(size_t)b * Ncols + n];
}

__global__ void maxout_k(const float* __restrict__ t, unsigned short* __restrict__ tm,
                         int Mh, int Mp, int N2) {
  int r = blockIdx.x;
  for (int m = threadIdx.x; m < Mh; m += blockDim.x) {
    float a = t[(size_t)r * N2 + 2 * m];
    float b = t[(size_t)r * N2 + 2 * m + 1];
    tm[(size_t)r * Mp + m] = f2bf(fmaxf(a, b));
  }
}

// ---------------------------------------------------------------------------
extern "C" void kernel_launch(void* const* d_in, const int* in_sizes, int n_in,
                              void* d_out, int out_size, void* d_ws, size_t ws_size,
                              hipStream_t stream) {
  (void)in_sizes; (void)n_in; (void)out_size;
  const int B = 32, S = 24, T = 24, V = 30000, E = 620, H = 1000, Mh = 500;
  const int Ep = 640, Hp = 1024, Mp = 512, M2 = 1000; // 2*Mh

  const int*   src     = (const int*)  d_in[0];
  const int*   tgt     = (const int*)  d_in[1];
  const float* src_emb = (const float*)d_in[3];
  const float* tgt_emb = (const float*)d_in[4];
  const float* enc_W  = (const float*)d_in[5],  *enc_Wz = (const float*)d_in[6],
             * enc_Wr = (const float*)d_in[7],  *enc_U  = (const float*)d_in[8],
             * enc_Uz = (const float*)d_in[9],  *enc_Ur = (const float*)d_in[10],
             * enc_b  = (const float*)d_in[11], *enc_bz = (const float*)d_in[12],
             * enc_br = (const float*)d_in[13];
  const float* dec_W  = (const float*)d_in[14], *dec_Wz = (const float*)d_in[15],
             * dec_Wr = (const float*)d_in[16], *dec_U  = (const float*)d_in[17],
             * dec_Uz = (const float*)d_in[18], *dec_Ur = (const float*)d_in[19],
             * dec_C  = (const float*)d_in[20], *dec_Cz = (const float*)d_in[21],
             * dec_Cr = (const float*)d_in[22], *dec_b  = (const float*)d_in[23],
             * dec_bz = (const float*)d_in[24], *dec_br = (const float*)d_in[25];
  const float* W_s = (const float*)d_in[26], *U_o = (const float*)d_in[27],
             * V_o = (const float*)d_in[28], *C_o = (const float*)d_in[29],
             * W_o = (const float*)d_in[30];
  float* out = (float*)d_out;

  // ---- workspace layout -------------------------------------------------
  char* ws = (char*)d_ws;
  size_t off = 0;
  auto alloc = [&](size_t bytes) -> void* {
    void* p = ws + off; off = (off + bytes + 255) & ~(size_t)255; return p;
  };
  typedef unsigned short u16;
  // zero-initialized zone (bf16 pads + initial states must be 0)
  u16*  h_bf   = (u16*) alloc((size_t)B * Hp * 2);           // encoder state bf16
  u16*  rh_bf  = (u16*) alloc((size_t)B * Hp * 2);
  u16*  s_bf   = (u16*) alloc((size_t)B * Hp * 2);           // decoder state bf16
  u16*  s_all  = (u16*) alloc((size_t)(T + 1) * B * Hp * 2); // s_t at rows t*B
  u16*  tmax   = (u16*) alloc((size_t)T * B * Mp * 2);
  float* h_f   = (float*)alloc((size_t)B * H * 4);           // encoder state f32 (h0=0)
  size_t zeroEnd = off;
  // filled-by-kernel zone
  u16* Xbf = (u16*)alloc((size_t)S * B * Ep * 2);
  u16* Ybf = (u16*)alloc((size_t)T * B * Ep * 2);
  u16 *encWb  = (u16*)alloc((size_t)H * Ep * 2),  *encWzb = (u16*)alloc((size_t)H * Ep * 2),
      *encWrb = (u16*)alloc((size_t)H * Ep * 2);
  u16 *encUb  = (u16*)alloc((size_t)H * Hp * 2),  *encUzb = (u16*)alloc((size_t)H * Hp * 2),
      *encUrb = (u16*)alloc((size_t)H * Hp * 2);
  u16 *decWb  = (u16*)alloc((size_t)H * Ep * 2),  *decWzb = (u16*)alloc((size_t)H * Ep * 2),
      *decWrb = (u16*)alloc((size_t)H * Ep * 2);
  u16 *decUb  = (u16*)alloc((size_t)H * Hp * 2),  *decUzb = (u16*)alloc((size_t)H * Hp * 2),
      *decUrb = (u16*)alloc((size_t)H * Hp * 2);
  u16 *decCb  = (u16*)alloc((size_t)H * Hp * 2),  *decCzb = (u16*)alloc((size_t)H * Hp * 2),
      *decCrb = (u16*)alloc((size_t)H * Hp * 2);
  u16 *Wsb = (u16*)alloc((size_t)H * Hp * 2);
  u16 *Uob = (u16*)alloc((size_t)M2 * Hp * 2);
  u16 *Vob = (u16*)alloc((size_t)M2 * Ep * 2);
  u16 *Cob = (u16*)alloc((size_t)M2 * Hp * 2);
  u16 *Wob = (u16*)alloc((size_t)V * Mp * 2);
  float *encPc = (float*)alloc((size_t)S * B * H * 4);   // x @ enc_W.T
  float *encPz = (float*)alloc((size_t)S * B * H * 4);
  float *encPr = (float*)alloc((size_t)S * B * H * 4);
  float *decPc = (float*)alloc((size_t)T * B * H * 4);
  float *decPz = (float*)alloc((size_t)T * B * H * 4);
  float *decPr = (float*)alloc((size_t)T * B * H * 4);
  float *yVo   = (float*)alloc((size_t)T * B * M2 * 4);
  float *cCr = (float*)alloc((size_t)B * H * 4), *cCz = (float*)alloc((size_t)B * H * 4);
  float *cC  = (float*)alloc((size_t)B * H * 4), *cCo = (float*)alloc((size_t)B * M2 * 4);
  float *z_f = (float*)alloc((size_t)B * H * 4);
  float *s_f = (float*)alloc((size_t)B * H * 4);
  float *tmp0 = (float*)alloc((size_t)B * H * 4);
  float *t_all = (float*)alloc((size_t)T * B * M2 * 4);
  if (off > ws_size) return;   // workspace too small: bail out safely

  // ---- 0) zero the pad/state zone ---------------------------------------
  zero_u32_k<<<1024, 256, 0, stream>>>((unsigned int*)ws, (long)(zeroEnd / 4));

  // ---- 1) embedding gathers (time-major rows r = s*B + b) ----------------
  gather_embed_k<<<S * B, 256, 0, stream>>>(src, src_emb, Xbf, B, S, E, Ep);
  gather_embed_k<<<T * B, 256, 0, stream>>>(tgt, tgt_emb, Ybf, B, T, E, Ep);

  // ---- 2) weight conversions f32 -> padded bf16 --------------------------
  struct Cvt { const float* w; u16* d; int O, K, Kp; };
  const Cvt cv[] = {
    {enc_W, encWb, H, E, Ep}, {enc_Wz, encWzb, H, E, Ep}, {enc_Wr, encWrb, H, E, Ep},
    {enc_U, encUb, H, H, Hp}, {enc_Uz, encUzb, H, H, Hp}, {enc_Ur, encUrb, H, H, Hp},
    {dec_W, decWb, H, E, Ep}, {dec_Wz, decWzb, H, E, Ep}, {dec_Wr, decWrb, H, E, Ep},
    {dec_U, decUb, H, H, Hp}, {dec_Uz, decUzb, H, H, Hp}, {dec_Ur, decUrb, H, H, Hp},
    {dec_C, decCb, H, H, Hp}, {dec_Cz, decCzb, H, H, Hp}, {dec_Cr, decCrb, H, H, Hp},
    {W_s, Wsb, H, H, Hp}, {U_o, Uob, M2, H, Hp}, {V_o, Vob, M2, E, Ep},
    {C_o, Cob, M2, H, Hp}, {W_o, Wob, V, Mh, Mp},
  };
  for (const Cvt& c : cv)
    cvt_bf16_k<<<c.O, 256, 0, stream>>>(c.w, c.d, c.K, c.Kp);

  // ---- 3) batched input projections (all timesteps at once) --------------
  dim3 gemmBlk(128);
  dim3 gProj((H + 127) / 128, (S * B + 31) / 32);
  gemm_bf16_k<<<gProj, gemmBlk, 0, stream>>>(Xbf, Ep, encWb,  Ep, H, encPc, H, S*B, H, Ep, nullptr, 0, 0, 0);
  gemm_bf16_k<<<gProj, gemmBlk, 0, stream>>>(Xbf, Ep, encWzb, Ep, H, encPz, H, S*B, H, Ep, nullptr, 0, 0, 0);
  gemm_bf16_k<<<gProj, gemmBlk, 0, stream>>>(Xbf, Ep, encWrb, Ep, H, encPr, H, S*B, H, Ep, nullptr, 0, 0, 0);
  gemm_bf16_k<<<gProj, gemmBlk, 0, stream>>>(Ybf, Ep, decWb,  Ep, H, decPc, H, T*B, H, Ep, nullptr, 0, 0, 0);
  gemm_bf16_k<<<gProj, gemmBlk, 0, stream>>>(Ybf, Ep, decWzb, Ep, H, decPz, H, T*B, H, Ep, nullptr, 0, 0, 0);
  gemm_bf16_k<<<gProj, gemmBlk, 0, stream>>>(Ybf, Ep, decWrb, Ep, H, decPr, H, T*B, H, Ep, nullptr, 0, 0, 0);
  gemm_bf16_k<<<gProj, gemmBlk, 0, stream>>>(Ybf, Ep, Vob,    Ep, M2, yVo,  M2, T*B, M2, Ep, nullptr, 0, 0, 0);

  // ---- 4) encoder recurrence (serial; 2 WMMA kernels per step) -----------
  dim3 gStep((H + 63) / 64);
  for (int s = 0; s < S; ++s) {
    const size_t o = (size_t)s * B * H;
    step_rz_k<<<gStep, 256, 0, stream>>>(h_bf, encUrb, encUzb, H, Hp,
                                         encPr + o, encPz + o, nullptr, nullptr,
                                         enc_br, enc_bz, h_f, rh_bf, z_f);
    step_upd_k<<<gStep, 256, 0, stream>>>(rh_bf, encUb, H, Hp, encPc + o, nullptr,
                                          enc_b, z_f, h_f, h_bf, nullptr);
  }

  // ---- 5) context projections + s0 ---------------------------------------
  dim3 gSmall((H + 127) / 128, 1);
  gemm_bf16_k<<<gSmall, gemmBlk, 0, stream>>>(h_bf, Hp, Wsb,    Hp, H,  tmp0, H,  B, H,  Hp, nullptr, 0, 0, 0);
  s0_tanh_k<<<B, 256, 0, stream>>>(tmp0, s_f, s_bf, s_all, H, Hp);
  gemm_bf16_k<<<gSmall, gemmBlk, 0, stream>>>(h_bf, Hp, decCrb, Hp, H,  cCr, H,  B, H,  Hp, nullptr, 0, 0, 0);
  gemm_bf16_k<<<gSmall, gemmBlk, 0, stream>>>(h_bf, Hp, decCzb, Hp, H,  cCz, H,  B, H,  Hp, nullptr, 0, 0, 0);
  gemm_bf16_k<<<gSmall, gemmBlk, 0, stream>>>(h_bf, Hp, decCb,  Hp, H,  cC,  H,  B, H,  Hp, nullptr, 0, 0, 0);
  gemm_bf16_k<<<gSmall, gemmBlk, 0, stream>>>(h_bf, Hp, Cob,    Hp, M2, cCo, M2, B, M2, Hp, nullptr, 0, 0, 0);

  // ---- 6) decoder recurrence; store s_t (bf16) for batched readout -------
  for (int t = 0; t < T; ++t) {
    const size_t o = (size_t)t * B * H;
    step_rz_k<<<gStep, 256, 0, stream>>>(s_bf, decUrb, decUzb, H, Hp,
                                         decPr + o, decPz + o, cCr, cCz,
                                         dec_br, dec_bz, s_f, rh_bf, z_f);
    step_upd_k<<<gStep, 256, 0, stream>>>(rh_bf, decUb, H, Hp, decPc + o, cC,
                                          dec_b, z_f, s_f, s_bf,
                                          s_all + (size_t)(t + 1) * B * Hp);
  }

  // ---- 7) batched maxout readout + logits --------------------------------
  add_bcast_k<<<T * B, 256, 0, stream>>>(yVo, cCo, B, M2);   // yVo += cCo[b]
  dim3 gT((M2 + 127) / 128, (T * B + 31) / 32);
  gemm_bf16_k<<<gT, gemmBlk, 0, stream>>>(s_all, Hp, Uob, Hp, M2, t_all, M2,
                                          T * B, M2, Hp, yVo, 0, 0, 0);
  maxout_k<<<T * B, 256, 0, stream>>>(t_all, tmax, Mh, Mp, M2);
  dim3 gLog((V + 127) / 128, (T * B + 31) / 32);
  gemm_bf16_k<<<gLog, gemmBlk, 0, stream>>>(tmax, Mp, Wob, Mp, V, out, V,
                                            T * B, V, Mp, nullptr, 1, B, T);
}
